// QuantizedColumnParallel_4492535791711
// MI455X (gfx1250) — matrix operand order repaired and checked by reading, not compiled, not run on hardware
//
#include <hip/hip_runtime.h>
#include <hip/hip_bf16.h>

// QuantizedColumnParallel: y[m, n] = (sum_k x[m,k] * wq[n,k]) * scale[n] + bias[n]
// M = B*S = 8192, K = 4096, N = 16384.
// int8-valued weights are exact in f16, so we run V_WMMA_F32_16X16X32_F16 with
// f32 accumulation and fold the per-channel dequant scale + bias into the epilogue.

typedef _Float16 v16h __attribute__((ext_vector_type(16)));
typedef _Float16 v8h  __attribute__((ext_vector_type(8)));
typedef float    v8f  __attribute__((ext_vector_type(8)));
typedef float    v4f  __attribute__((ext_vector_type(4)));
typedef int      v4i  __attribute__((ext_vector_type(4)));

#define M_DIM 8192
#define N_DIM 16384
#define K_DIM 4096

#define BM 128
#define BN 128
#define BK 32
#define LDSS 40          // LDS row stride in halves (BK + 8 pad) -> 80B rows, 16B aligned
#define NTHREADS 256

// Convert 16 fp32 (A) and 16 int32 (B) lane-local elements to f16 and store to LDS.
__device__ __forceinline__ void stage_tiles(_Float16* dstA, _Float16* dstB,
                                            v4f a0, v4f a1, v4f a2, v4f a3,
                                            v4i b0, v4i b1, v4i b2, v4i b3)
{
    v8h h0, h1, g0, g1;
#pragma unroll
    for (int j = 0; j < 4; ++j) {
        h0[j]     = (_Float16)a0[j];
        h0[j + 4] = (_Float16)a1[j];
        h1[j]     = (_Float16)a2[j];
        h1[j + 4] = (_Float16)a3[j];
        g0[j]     = (_Float16)(float)b0[j];   // int8-valued -> exact in f16
        g0[j + 4] = (_Float16)(float)b1[j];
        g1[j]     = (_Float16)(float)b2[j];
        g1[j + 4] = (_Float16)(float)b3[j];
    }
    *(v8h*)(dstA)     = h0;
    *(v8h*)(dstA + 8) = h1;
    *(v8h*)(dstB)     = g0;
    *(v8h*)(dstB + 8) = g1;
}

__global__ __launch_bounds__(NTHREADS)
void qcp_wmma_f16_kernel(const float* __restrict__ x,
                         const int*   __restrict__ wq,
                         const float* __restrict__ scale,
                         const float* __restrict__ bias,
                         float*       __restrict__ out)
{
    __shared__ __align__(16) _Float16 As[2][BM * LDSS];
    __shared__ __align__(16) _Float16 Bs[2][BN * LDSS];

    const int tid  = threadIdx.x;
    const int lane = tid & 31;
    const int wave = tid >> 5;

    const int bm = blockIdx.x * BM;   // grid.x runs over M -> weight strip reused in L2
    const int bn = blockIdx.y * BN;

    // 8 waves: 2 along M (64 rows), 4 along N (32 cols)
    const int wm = (wave & 1) * 64;
    const int wn = (wave >> 1) * 32;

    // fragment lane decomposition (wave32)
    const int fi   = lane & 15;
    const int half = lane >> 4;

    // global staging: each thread owns 16 contiguous K-elements of one tile row
    const int ldr = tid >> 1;          // 0..127
    const int ldk = (tid & 1) * 16;    // 0 or 16

    const float* xg = x  + (size_t)(bm + ldr) * K_DIM + ldk;
    const int*   wg = wq + (size_t)(bn + ldr) * K_DIM + ldk;

    v8f acc[4][2];
#pragma unroll
    for (int mt = 0; mt < 4; ++mt)
#pragma unroll
        for (int nt = 0; nt < 2; ++nt)
            acc[mt][nt] = (v8f){0.f, 0.f, 0.f, 0.f, 0.f, 0.f, 0.f, 0.f};

    const int NK = K_DIM / BK;   // 128 K-steps

    // ---- prologue: stage tile 0 into buffer 0 ----
    {
        v4f a0 = *(const v4f*)(xg + 0);
        v4f a1 = *(const v4f*)(xg + 4);
        v4f a2 = *(const v4f*)(xg + 8);
        v4f a3 = *(const v4f*)(xg + 12);
        v4i b0 = __builtin_nontemporal_load((const v4i*)(wg + 0));
        v4i b1 = __builtin_nontemporal_load((const v4i*)(wg + 4));
        v4i b2 = __builtin_nontemporal_load((const v4i*)(wg + 8));
        v4i b3 = __builtin_nontemporal_load((const v4i*)(wg + 12));
        stage_tiles(&As[0][ldr * LDSS + ldk], &Bs[0][ldr * LDSS + ldk],
                    a0, a1, a2, a3, b0, b1, b2, b3);
    }
    __syncthreads();

    // ---- main loop: double-buffered LDS, register prefetch of next tile ----
    for (int kt = 0; kt < NK; ++kt) {
        const int  cur  = kt & 1;
        const int  nxt  = cur ^ 1;
        const bool more = (kt + 1) < NK;

        v4f a0 = {}, a1 = {}, a2 = {}, a3 = {};
        v4i b0 = {}, b1 = {}, b2 = {}, b3 = {};
        if (more) {
            const float* pa = xg + (size_t)(kt + 1) * BK;
            const int*   pb = wg + (size_t)(kt + 1) * BK;
            a0 = *(const v4f*)(pa + 0);
            a1 = *(const v4f*)(pa + 4);
            a2 = *(const v4f*)(pa + 8);
            a3 = *(const v4f*)(pa + 12);
            b0 = __builtin_nontemporal_load((const v4i*)(pb + 0));
            b1 = __builtin_nontemporal_load((const v4i*)(pb + 4));
            b2 = __builtin_nontemporal_load((const v4i*)(pb + 8));
            b3 = __builtin_nontemporal_load((const v4i*)(pb + 12));
        }

        // A fragments: 16x32 f16.  Lane holds row fi, K in [half*8, half*8+8) and [16+half*8, ...)
        v16h af[4];
#pragma unroll
        for (int mt = 0; mt < 4; ++mt) {
            const _Float16* base = &As[cur][(wm + mt * 16 + fi) * LDSS];
            v8h lo = *(const v8h*)(base + half * 8);
            v8h hi = *(const v8h*)(base + 16 + half * 8);
            af[mt] = __builtin_shufflevector(lo, hi, 0, 1, 2, 3, 4, 5, 6, 7,
                                             8, 9, 10, 11, 12, 13, 14, 15);
        }
        // B fragments: 32x16 f16 (B = W^T).  Lane holds W-row fi, K in [half*8, half*8+8) and [16+half*8, ...)
        v16h bf[2];
#pragma unroll
        for (int nt = 0; nt < 2; ++nt) {
            const _Float16* base = &Bs[cur][(wn + nt * 16 + fi) * LDSS];
            v8h lo = *(const v8h*)(base + half * 8);
            v8h hi = *(const v8h*)(base + 16 + half * 8);
            bf[nt] = __builtin_shufflevector(lo, hi, 0, 1, 2, 3, 4, 5, 6, 7,
                                             8, 9, 10, 11, 12, 13, 14, 15);
        }

        // 8 WMMAs per wave per K-step: 64x32 output sub-tile
#pragma unroll
        for (int mt = 0; mt < 4; ++mt)
#pragma unroll
            for (int nt = 0; nt < 2; ++nt)
                acc[mt][nt] = __builtin_amdgcn_wmma_f32_16x16x32_f16(
                    false, af[mt], false, bf[nt], (short)0, acc[mt][nt], false, false);

        if (more)
            stage_tiles(&As[nxt][ldr * LDSS + ldk], &Bs[nxt][ldr * LDSS + ldk],
                        a0, a1, a2, a3, b0, b1, b2, b3);
        __syncthreads();
    }

    // ---- epilogue: dequant scale + bias, streaming (non-temporal) stores ----
#pragma unroll
    for (int nt = 0; nt < 2; ++nt) {
        const int   ng = bn + wn + nt * 16 + fi;
        const float s  = scale[ng];
        const float bb = bias[ng];
#pragma unroll
        for (int mt = 0; mt < 4; ++mt) {
            const int mg0 = bm + wm + mt * 16 + half * 8;
#pragma unroll
            for (int v = 0; v < 8; ++v) {
                const float r = acc[mt][nt][v] * s + bb;
                __builtin_nontemporal_store(r, out + (size_t)(mg0 + v) * N_DIM + ng);
            }
        }
    }
}

extern "C" void kernel_launch(void* const* d_in, const int* in_sizes, int n_in,
                              void* d_out, int out_size, void* d_ws, size_t ws_size,
                              hipStream_t stream) {
    const float* x     = (const float*)d_in[0];
    const int*   wq    = (const int*)d_in[1];
    const float* scale = (const float*)d_in[2];
    const float* bias  = (const float*)d_in[3];
    float*       out   = (float*)d_out;

    dim3 grid(M_DIM / BM, N_DIM / BN);   // (64, 128): M fastest -> weight-strip L2 reuse
    qcp_wmma_f16_kernel<<<grid, NTHREADS, 0, stream>>>(x, wq, scale, bias, out);
}
